// MoE_Cute_54580444398294
// MI455X (gfx1250) — matrix operand order repaired and checked by reading, not compiled, not run on hardware
//
#include <hip/hip_runtime.h>

typedef __bf16 bf16;
typedef __attribute__((ext_vector_type(8)))  bf16  v8bf;
typedef __attribute__((ext_vector_type(16))) bf16  v16bf;
typedef __attribute__((ext_vector_type(8)))  float v8f;
typedef __attribute__((ext_vector_type(4)))  unsigned int v4u;
typedef __attribute__((ext_vector_type(8)))  int v8i;
typedef __attribute__((ext_vector_type(4)))  int v4i;

constexpr int T_ = 8192;   // tokens (4*2048)
constexpr int H_ = 2048;   // hidden
constexpr int I_ = 4096;   // intermediate
constexpr int E_ = 8;      // experts
constexpr int BM = 128, BN = 128, BK = 32;
constexpr int LDP = 40;    // LDS pitch (bf16 elems): 80B rows, 16B aligned, conflict-free

// ---------------- router: logits, top-2 softmax, per-expert counts ----------------
__global__ __launch_bounds__(256) void k_router(const float* __restrict__ x,
                                                const float* __restrict__ gw,
                                                float* __restrict__ logits,
                                                int* __restrict__ topk_id,
                                                float* __restrict__ topk_w,
                                                int* __restrict__ cnt) {
  const int wid  = threadIdx.x >> 5;
  const int lane = threadIdx.x & 31;
  const int t    = blockIdx.x * 8 + wid;
  float acc[E_];
#pragma unroll
  for (int e = 0; e < E_; ++e) acc[e] = 0.0f;
  const float* xr = x + (size_t)t * H_;
  for (int it = 0; it < H_ / 128; ++it) {
    const int k = it * 128 + lane * 4;
    const float4 xv = *(const float4*)(xr + k);
#pragma unroll
    for (int e = 0; e < E_; ++e) {
      const float4 gv = *(const float4*)(gw + e * H_ + k);
      acc[e] = fmaf(xv.x, gv.x, fmaf(xv.y, gv.y, fmaf(xv.z, gv.z, fmaf(xv.w, gv.w, acc[e]))));
    }
  }
#pragma unroll
  for (int e = 0; e < E_; ++e)
    for (int off = 16; off > 0; off >>= 1) acc[e] += __shfl_xor(acc[e], off, 32);
  if (lane < E_) logits[(size_t)t * E_ + lane] = acc[lane];
  if (lane == 0) {
    float best = -3.4e38f, sec = -3.4e38f; int bi = 0, si = 0;
#pragma unroll
    for (int e = 0; e < E_; ++e) {
      const float v = acc[e];
      if (v > best) { sec = best; si = bi; best = v; bi = e; }
      else if (v > sec) { sec = v; si = e; }
    }
    const float p  = __expf(sec - best);
    const float w0 = 1.0f / (1.0f + p);
    topk_id[2 * t] = bi; topk_id[2 * t + 1] = si;
    topk_w[2 * t]  = w0; topk_w[2 * t + 1]  = 1.0f - w0;
    atomicAdd(&cnt[bi], 1); atomicAdd(&cnt[si], 1);
  }
}

__global__ void k_offsets(const int* __restrict__ cnt, int* __restrict__ offs) {
  if (threadIdx.x == 0) { int r = 0; for (int e = 0; e < E_; ++e) { offs[e] = r; r += cnt[e]; } }
}

// ------------- deterministic per-expert compaction (ballot + block scan) -------------
__global__ __launch_bounds__(256) void k_build(const int* __restrict__ topk_id,
                                               const float* __restrict__ topk_w,
                                               const int* __restrict__ offs,
                                               int* __restrict__ rowmap,
                                               float* __restrict__ rowgate) {
  const int e = blockIdx.x;
  const int tid = threadIdx.x, lane = tid & 31, wid = tid >> 5;
  __shared__ int wsum[8];
  int running = offs[e];
  for (int base = 0; base < T_; base += 256) {
    const int t  = base + tid;
    const int i0 = topk_id[2 * t], i1 = topk_id[2 * t + 1];
    const int flag = (i0 == e) || (i1 == e);
    const float w  = (i0 == e) ? topk_w[2 * t] : topk_w[2 * t + 1];
    const unsigned mask = (unsigned)__ballot(flag);
    const int lanePre = __popc(mask & ((1u << lane) - 1u));
    if (lane == 0) wsum[wid] = __popc(mask);
    __syncthreads();
    int wPre = 0, bTot = 0;
#pragma unroll
    for (int ww = 0; ww < 8; ++ww) { const int c = wsum[ww]; if (ww < wid) wPre += c; bTot += c; }
    if (flag) { const int s = running + wPre + lanePre; rowmap[s] = t; rowgate[s] = w; }
    running += bTot;
    __syncthreads();
  }
}

// ---------------- shared WMMA helpers ----------------
__device__ inline v8bf cv8(const float4 a, const float4 b) {
  v8bf r;
  r[0] = (bf16)a.x; r[1] = (bf16)a.y; r[2] = (bf16)a.z; r[3] = (bf16)a.w;
  r[4] = (bf16)b.x; r[5] = (bf16)b.y; r[6] = (bf16)b.z; r[7] = (bf16)b.w;
  return r;
}
__device__ inline v16bf joinv(const v8bf lo, const v8bf hi) {
  return __builtin_shufflevector(lo, hi, 0,1,2,3,4,5,6,7,8,9,10,11,12,13,14,15);
}

// ----- TDM: load a [128 rows x 32 bf16] tile into LDS with 16B/row padding -----
// rows stride = I_ elems; rows >= rows_valid are OOB -> zero-filled by TDM.
__device__ inline void tdm_load_tile(unsigned lds_byte, unsigned long long gaddr,
                                     unsigned rows_valid) {
  v4u g0;
  g0[0] = 1u;                                              // count=1, user descriptor
  g0[1] = lds_byte;                                        // lds_addr
  g0[2] = (unsigned)(gaddr & 0xffffffffu);                 // global_addr[31:0]
  g0[3] = (unsigned)((gaddr >> 32) & 0x01ffffffu) | (2u << 30);  // global_addr[56:32] | type=2
  v8i g1;
  // data_size=2B | pad_enable | pad_interval: 16 dwords (code 3) | pad_amount: 4 dwords (code 3)
  g1[0] = (int)((1u << 16) | (1u << 20) | (3u << 22) | (3u << 25));
  g1[1] = (int)(((unsigned)I_ & 0xffffu) << 16);           // tensor_dim0[15:0] = 4096
  g1[2] = (int)((rows_valid & 0xffffu) << 16);             // tensor_dim0[31:16]=0 | tensor_dim1[15:0]
  g1[3] = (int)((rows_valid >> 16) & 0xffffu) | (int)(32u << 16); // dim1[31:16] | tile_dim0=32
  g1[4] = 128;                                             // tile_dim1=128, tile_dim2=0
  g1[5] = (int)I_;                                         // tensor_dim0_stride[31:0] = 4096
  g1[6] = 0;                                               // stride hi | dim1_stride lo (unused, 2D)
  g1[7] = 0;
  const v4i z4 = {0, 0, 0, 0};
#if __clang_major__ >= 23
  const v8i z8 = {0, 0, 0, 0, 0, 0, 0, 0};
  __builtin_amdgcn_tensor_load_to_lds(g0, g1, z4, z4, z8, 0);
#else
  __builtin_amdgcn_tensor_load_to_lds(g0, g1, z4, z4, 0);
#endif
}

// ---------------- fc1: h = gelu(x @ wfc^T + bfc), bf16 out ----------------
__global__ __launch_bounds__(256) void k_fc1(const float* __restrict__ x,
                                             const float* __restrict__ wfc,
                                             const float* __restrict__ bfc,
                                             const int* __restrict__ rowmap,
                                             const int* __restrict__ cnt,
                                             const int* __restrict__ offs,
                                             bf16* __restrict__ hbuf) {
  const int e    = blockIdx.z;
  const int m0   = blockIdx.y * BM;
  const int n0   = blockIdx.x * BN;
  const int mcnt = cnt[e];
  if (m0 >= mcnt) return;
  const int base = offs[e];

  __shared__ bf16 As[2][BM * LDP];
  __shared__ bf16 Bs[2][BN * LDP];

  const int tid    = threadIdx.x;
  const int lrow   = tid >> 1;
  const int lchunk = (tid & 1) * 16;
  const int arow   = m0 + lrow;
  const int tok    = rowmap[base + ((arow < mcnt) ? arow : 0)];
  const float* aptr = x   + (size_t)tok * H_ + lchunk;
  const float* bptr = wfc + ((size_t)e * I_ + (n0 + lrow)) * H_ + lchunk;

  const int lane = tid & 31, wid = tid >> 5;
  const int waveM = wid >> 2, waveN = wid & 3;
  const int half = lane >> 4, nm = lane & 15;

  float4 sa[4], sb[4];
  auto gload = [&](int kt) {
    const float4* ap = (const float4*)(aptr + (size_t)kt * BK);
    const float4* bp = (const float4*)(bptr + (size_t)kt * BK);
#pragma unroll
    for (int i = 0; i < 4; ++i) { sa[i] = ap[i]; sb[i] = bp[i]; }
  };
  auto sstore = [&](int buf) {
    bf16* a = &As[buf][lrow * LDP + lchunk];
    bf16* b = &Bs[buf][lrow * LDP + lchunk];
    *(v8bf*)(a)     = cv8(sa[0], sa[1]);  *(v8bf*)(a + 8) = cv8(sa[2], sa[3]);
    *(v8bf*)(b)     = cv8(sb[0], sb[1]);  *(v8bf*)(b + 8) = cv8(sb[2], sb[3]);
  };

  v8f acc[4][2];
  const v8f z8 = {0.f,0.f,0.f,0.f,0.f,0.f,0.f,0.f};
#pragma unroll
  for (int mi = 0; mi < 4; ++mi)
#pragma unroll
    for (int ni = 0; ni < 2; ++ni) acc[mi][ni] = z8;

  gload(0); sstore(0); __syncthreads();
  const int KT = H_ / BK;
  for (int kt = 0; kt < KT; ++kt) {
    const int cur = kt & 1;
    if (kt + 1 < KT) gload(kt + 1);
    if (kt + 2 < KT) {
      __builtin_prefetch((const void*)(bptr + (size_t)(kt + 2) * BK), 0, 1);
      __builtin_prefetch((const void*)(aptr + (size_t)(kt + 2) * BK), 0, 1);
    }
    const bf16* pa = &As[cur][(waveM * 64 + nm) * LDP + 8 * half];
    const bf16* pb = &Bs[cur][(waveN * 32 + nm) * LDP + 8 * half];
    v16bf bfr[2];
#pragma unroll
    for (int ni = 0; ni < 2; ++ni)
      bfr[ni] = joinv(*(const v8bf*)(pb + ni * 16 * LDP), *(const v8bf*)(pb + ni * 16 * LDP + 16));
#pragma unroll
    for (int mi = 0; mi < 4; ++mi) {
      const v16bf afr = joinv(*(const v8bf*)(pa + mi * 16 * LDP), *(const v8bf*)(pa + mi * 16 * LDP + 16));
#pragma unroll
      for (int ni = 0; ni < 2; ++ni)
        acc[mi][ni] = __builtin_amdgcn_wmma_f32_16x16x32_bf16(
            false, afr, false, bfr[ni], (short)0, acc[mi][ni], false, false);
    }
    if (kt + 1 < KT) sstore(1 - cur);
    __syncthreads();
  }

  // epilogue: bias + exact GELU -> bf16 h rows (by slot, contiguous per expert)
#pragma unroll
  for (int mi = 0; mi < 4; ++mi) {
#pragma unroll
    for (int i = 0; i < 8; ++i) {
      const int mrow = m0 + waveM * 64 + mi * 16 + half * 8 + i;
      if (mrow < mcnt) {
#pragma unroll
        for (int ni = 0; ni < 2; ++ni) {
          const int col = n0 + waveN * 32 + ni * 16 + nm;
          const float v = acc[mi][ni][i] + bfc[e * I_ + col];
          const float g = 0.5f * v * (1.0f + erff(v * 0.70710678118654752f));
          hbuf[(size_t)(base + mrow) * I_ + col] = (bf16)g;
        }
      }
    }
  }
}

// ------- fc2: out[tok] += gate * (h @ wproj^T + bproj); A tiles via TDM -------
__global__ __launch_bounds__(256) void k_fc2(const bf16* __restrict__ hbuf,
                                             const float* __restrict__ wproj,
                                             const float* __restrict__ bproj,
                                             const int* __restrict__ rowmap,
                                             const float* __restrict__ rowgate,
                                             const int* __restrict__ cnt,
                                             const int* __restrict__ offs,
                                             float* __restrict__ out) {
  const int e    = blockIdx.z;
  const int m0   = blockIdx.y * BM;
  const int n0   = blockIdx.x * BN;
  const int mcnt = cnt[e];
  if (m0 >= mcnt) return;
  const int base = offs[e];

  __shared__ bf16 As[2][BM * LDP];
  __shared__ bf16 Bs[2][BN * LDP];

  const int tid    = threadIdx.x;
  const int lrow   = tid >> 1;
  const int lchunk = (tid & 1) * 16;
  const float* bptr = wproj + ((size_t)e * H_ + (n0 + lrow)) * I_ + lchunk;

  const int lane = tid & 31, wid = tid >> 5;
  const int waveM = wid >> 2, waveN = wid & 3;
  const int half = lane >> 4, nm = lane & 15;

  // TDM source: contiguous slot rows of h; ragged tail zero-filled via tensor_dim1
  const unsigned rows_valid = (unsigned)(mcnt - m0);
  const unsigned long long ga0 =
      (unsigned long long)(size_t)(hbuf + (size_t)(base + m0) * I_);
  const unsigned ldsA[2] = { (unsigned)(size_t)(void*)&As[0][0],
                             (unsigned)(size_t)(void*)&As[1][0] };

  float4 sb[4];
  auto gloadB = [&](int kt) {
    const float4* bp = (const float4*)(bptr + (size_t)kt * BK);
#pragma unroll
    for (int i = 0; i < 4; ++i) sb[i] = bp[i];
  };
  auto sstoreB = [&](int buf) {
    bf16* b = &Bs[buf][lrow * LDP + lchunk];
    *(v8bf*)(b)     = cv8(sb[0], sb[1]);
    *(v8bf*)(b + 8) = cv8(sb[2], sb[3]);
  };

  v8f acc[4][2];
  const v8f z8 = {0.f,0.f,0.f,0.f,0.f,0.f,0.f,0.f};
#pragma unroll
  for (int mi = 0; mi < 4; ++mi)
#pragma unroll
    for (int ni = 0; ni < 2; ++ni) acc[mi][ni] = z8;

  if (wid == 0) tdm_load_tile(ldsA[0], ga0, rows_valid);
  gloadB(0); sstoreB(0);
  if (wid == 0) __builtin_amdgcn_s_wait_tensorcnt(0);
  __syncthreads();

  const int KT = I_ / BK;
  for (int kt = 0; kt < KT; ++kt) {
    const int cur = kt & 1;
    if (kt + 1 < KT) {
      if (wid == 0)
        tdm_load_tile(ldsA[1 - cur], ga0 + (unsigned long long)(kt + 1) * (BK * 2), rows_valid);
      gloadB(kt + 1);
    }
    if (kt + 2 < KT) __builtin_prefetch((const void*)(bptr + (size_t)(kt + 2) * BK), 0, 1);
    const bf16* pa = &As[cur][(waveM * 64 + nm) * LDP + 8 * half];
    const bf16* pb = &Bs[cur][(waveN * 32 + nm) * LDP + 8 * half];
    v16bf bfr[2];
#pragma unroll
    for (int ni = 0; ni < 2; ++ni)
      bfr[ni] = joinv(*(const v8bf*)(pb + ni * 16 * LDP), *(const v8bf*)(pb + ni * 16 * LDP + 16));
#pragma unroll
    for (int mi = 0; mi < 4; ++mi) {
      const v16bf afr = joinv(*(const v8bf*)(pa + mi * 16 * LDP), *(const v8bf*)(pa + mi * 16 * LDP + 16));
#pragma unroll
      for (int ni = 0; ni < 2; ++ni)
        acc[mi][ni] = __builtin_amdgcn_wmma_f32_16x16x32_bf16(
            false, afr, false, bfr[ni], (short)0, acc[mi][ni], false, false);
    }
    if (kt + 1 < KT) sstoreB(1 - cur);
    if (wid == 0) __builtin_amdgcn_s_wait_tensorcnt(0);
    __syncthreads();
  }

  // epilogue: gate * (y + bproj) scatter-accumulated (exactly 2 adds per element -> order-invariant)
#pragma unroll
  for (int mi = 0; mi < 4; ++mi) {
#pragma unroll
    for (int i = 0; i < 8; ++i) {
      const int mrow = m0 + waveM * 64 + mi * 16 + half * 8 + i;
      if (mrow < mcnt) {
        const int slot = base + mrow;
        const int tok  = rowmap[slot];
        const float g  = rowgate[slot];
#pragma unroll
        for (int ni = 0; ni < 2; ++ni) {
          const int col = n0 + waveN * 32 + ni * 16 + nm;
          const float v = acc[mi][ni][i] + bproj[e * H_ + col];
          atomicAdd(&out[(size_t)tok * H_ + col], g * v);
        }
      }
    }
  }
}

// ---------------- host launcher ----------------
extern "C" void kernel_launch(void* const* d_in, const int* in_sizes, int n_in,
                              void* d_out, int out_size, void* d_ws, size_t ws_size,
                              hipStream_t stream) {
  const float* x     = (const float*)d_in[0];
  const float* gw    = (const float*)d_in[1];
  const float* wfc   = (const float*)d_in[2];
  const float* bfc   = (const float*)d_in[3];
  const float* wproj = (const float*)d_in[4];
  const float* bproj = (const float*)d_in[5];

  float* out    = (float*)d_out;
  float* logits = out + (size_t)T_ * H_;

  char* ws = (char*)d_ws;
  int*   cnt     = (int*)(ws + 0);
  int*   offs    = (int*)(ws + 64);
  int*   topk_id = (int*)(ws + 256);
  float* topk_w  = (float*)(ws + 256 + 65536);
  int*   rowmap  = (int*)(ws + 256 + 131072);
  float* rowgate = (float*)(ws + 256 + 196608);
  bf16*  hbuf    = (bf16*)(ws + 262400);   // 2T x I bf16 = 128 MB (L2-resident for fc2)

  hipMemsetAsync(cnt, 0, 256, stream);
  hipMemsetAsync(out, 0, (size_t)T_ * H_ * sizeof(float), stream);

  k_router<<<T_ / 8, 256, 0, stream>>>(x, gw, logits, topk_id, topk_w, cnt);
  k_offsets<<<1, 32, 0, stream>>>(cnt, offs);
  k_build<<<E_, 256, 0, stream>>>(topk_id, topk_w, offs, rowmap, rowgate);

  dim3 g1(I_ / BN, T_ / BM, E_);   // (32, 64, 8)
  k_fc1<<<g1, 256, 0, stream>>>(x, wfc, bfc, rowmap, cnt, offs, hbuf);

  dim3 g2(H_ / BN, T_ / BM, E_);   // (16, 64, 8)
  k_fc2<<<g2, 256, 0, stream>>>(hbuf, wproj, bproj, rowmap, rowgate, cnt, offs, out);
}
